// Net_56221121905186
// MI455X (gfx1250) — matrix operand order repaired and checked by reading, not compiled, not run on hardware
//
#include <hip/hip_runtime.h>
#include <cmath>

// Problem constants (must match reference)
#define NN  100000
#define NP  100032          // NN padded to 64-row multiple (workspace matrices)
#define FF  512
#define HH  256
#define CLS 40
#define EE  3200000
#define LL  4

#define LDSK 36   // padded K-stride (bf16 elems): 18 dwords -> conflict-free frag gathers

typedef __attribute__((ext_vector_type(16))) __bf16 v16bf;
typedef __attribute__((ext_vector_type(8)))  float  v8f;
typedef __attribute__((ext_vector_type(8)))  unsigned short u16x8;
typedef __attribute__((ext_vector_type(2)))  unsigned uint2v;

union FragAB { v16bf v; unsigned u[8]; };
union FragC  { v8f   v; float    f[8]; };

__device__ __forceinline__ unsigned short f2bf(float f) {
  unsigned u = __float_as_uint(f);
  unsigned r = (u + 0x7FFFu + ((u >> 16) & 1u)) >> 16;   // round-to-nearest-even
  return (unsigned short)r;
}
__device__ __forceinline__ unsigned pack2bf(float a, float b) {
  return (unsigned)f2bf(a) | ((unsigned)f2bf(b) << 16);
}

__device__ __forceinline__ float uaf(float x, float A, float B, float C, float D, float Ee) {
  float p1 = A * (x + B) + fminf(fmaxf(C * x * x, -100.f), 100.f);
  float p2 = D * (x - B);
  float sp1 = fmaxf(p1, 0.f) + log1pf(expf(-fabsf(p1)));
  float sp2 = fmaxf(p2, 0.f) + log1pf(expf(-fabsf(p2)));
  return sp1 - sp2 + Ee;
}

// ---------------------------------------------------------------------------
// Weight conversion fp32 -> bf16 (once per launch)
// ---------------------------------------------------------------------------
__global__ void __launch_bounds__(256)
k_cvt(const float* __restrict__ W1, const float* __restrict__ Wc,
      unsigned short* __restrict__ W1bf, unsigned short* __restrict__ Wcbf) {
  int i = blockIdx.x * 256 + threadIdx.x;
  if (i < FF * HH) W1bf[i] = f2bf(W1[i]);
  if (i < LL * HH * HH) Wcbf[i] = f2bf(Wc[i]);
}

// ---------------------------------------------------------------------------
// Tiled WMMA GEMM (bf16 in, f32 accum), 64x64 block tile, 8 waves, LDS
// double-buffered software pipeline (1 barrier / K-step).
// MODE 0:  out = uaf(A@B + bias); duplicates result into outX0 (x0 = h).
//          A = x (exact-size input) -> branchless clamp+select row guard.
// MODE 1:  S = 0.9*Aa + 0.1*Ab staged on the fly (padded ws, no guards);
//          out = uaf((1-beta)*S + beta*(S@B))
// A-stride = K. B is K x HH row-major bf16.
// ---------------------------------------------------------------------------
template <int MODE>
__global__ void __launch_bounds__(256)
k_gemm(const float* __restrict__ Aa, const float* __restrict__ Ab,
       const unsigned short* __restrict__ Bbf, const float* __restrict__ bias,
       float* __restrict__ outH, float* __restrict__ outX0,
       int K, float beta,
       const float* pA, const float* pB, const float* pC,
       const float* pD, const float* pE) {
  __shared__ __align__(16) unsigned short ldsA[2][64 * LDSK];  // [m][k]
  __shared__ __align__(16) unsigned short ldsB[2][64 * LDSK];  // [n][k]

  const int tid   = threadIdx.x;
  const int wid   = tid >> 5;
  const int lane  = tid & 31;
  const int waveM = wid >> 1;       // 0..3 -> 16-row strip
  const int waveN = wid & 1;        // 0..1 -> 32-col strip
  const int mBase = blockIdx.x * 64;
  const int nBase = blockIdx.y * 64;
  const int half  = lane >> 4;

  // --- staging geometry: A = 2 float4 chunks/thread, B = one b128/thread
  const int ar0 = tid >> 3,          ac0 = (tid & 7) * 4;          // chunk tid
  const int ar1 = (tid + 256) >> 3,  ac1 = (tid & 7) * 4;          // chunk tid+256
  const int bk  = tid >> 3,          bn  = (tid & 7) * 8;          // B: k-row, n-col
  int rowA0 = mBase + ar0, rowA1 = mBase + ar1;
  bool v0 = true, v1 = true;
  if (MODE == 0) {                          // branchless guard vs exact-size input
    v0 = rowA0 < NN; v1 = rowA1 < NN;
    rowA0 = v0 ? rowA0 : (NN - 1);
    rowA1 = v1 ? rowA1 : (NN - 1);
  }

  float4 fa0, fa1, fs0, fs1;   // prefetch regs (fs* = second source, MODE 1)
  u16x8  fb;

  auto prefetch = [&](int s) {
    const int kk = s * 32;
    fa0 = *(const float4*)(Aa + (size_t)rowA0 * K + kk + ac0);
    fa1 = *(const float4*)(Aa + (size_t)rowA1 * K + kk + ac1);
    if (MODE == 1) {
      fs0 = *(const float4*)(Ab + (size_t)rowA0 * K + kk + ac0);
      fs1 = *(const float4*)(Ab + (size_t)rowA1 * K + kk + ac1);
    }
    fb = *(const u16x8*)(Bbf + (size_t)(kk + bk) * HH + nBase + bn);
  };

  auto stage = [&](int buf) {
    float4 s0 = fa0, s1 = fa1;
    if (MODE == 1) {
      s0.x = 0.9f * fa0.x + 0.1f * fs0.x;  s0.y = 0.9f * fa0.y + 0.1f * fs0.y;
      s0.z = 0.9f * fa0.z + 0.1f * fs0.z;  s0.w = 0.9f * fa0.w + 0.1f * fs0.w;
      s1.x = 0.9f * fa1.x + 0.1f * fs1.x;  s1.y = 0.9f * fa1.y + 0.1f * fs1.y;
      s1.z = 0.9f * fa1.z + 0.1f * fs1.z;  s1.w = 0.9f * fa1.w + 0.1f * fs1.w;
    }
    if (MODE == 0) {
      if (!v0) { s0.x = s0.y = s0.z = s0.w = 0.f; }
      if (!v1) { s1.x = s1.y = s1.z = s1.w = 0.f; }
    }
    uint2v p0; p0.x = pack2bf(s0.x, s0.y); p0.y = pack2bf(s0.z, s0.w);
    uint2v p1; p1.x = pack2bf(s1.x, s1.y); p1.y = pack2bf(s1.z, s1.w);
    *(uint2v*)&ldsA[buf][ar0 * LDSK + ac0] = p0;
    *(uint2v*)&ldsA[buf][ar1 * LDSK + ac1] = p1;
#pragma unroll
    for (int j = 0; j < 8; j++) ldsB[buf][(bn + j) * LDSK + bk] = fb[j];
  };

  FragC acc0, acc1;
#pragma unroll
  for (int i = 0; i < 8; i++) { acc0.f[i] = 0.f; acc1.f[i] = 0.f; }

  const int nsteps = K >> 5;
  prefetch(0);
  for (int s = 0; s < nsteps; ++s) {
    const int buf = s & 1;
    stage(buf);
    __syncthreads();
    if (s + 1 < nsteps) prefetch(s + 1);   // overlap next tile's loads with WMMA

    // A fragment: 16x32 bf16 (lanes 0-15: K 0..7,16..23 ; lanes 16-31: +8)
    FragAB afrag;
    {
      const int m = waveM * 16 + (lane & 15);
#pragma unroll
      for (int j = 0; j < 8; j++) {
        int kk = (j & 3) * 2 + (j >> 2) * 16 + half * 8;
        afrag.u[j] = *(const unsigned*)&ldsA[buf][m * LDSK + kk];
      }
    }
#pragma unroll
    for (int t = 0; t < 2; t++) {
      FragAB bfrag;
      const int n = waveN * 32 + t * 16 + (lane & 15);
#pragma unroll
      for (int j = 0; j < 8; j++) {
        int kk = (j & 3) * 2 + (j >> 2) * 16 + half * 8;        // lanes 0-15: K 0..7,16..23 ; 16-31: +8
        bfrag.u[j] = *(const unsigned*)&ldsB[buf][n * LDSK + kk];
      }
      FragC& acc = t ? acc1 : acc0;
      acc.v = __builtin_amdgcn_wmma_f32_16x16x32_bf16(
          false, afrag.v, false, bfrag.v, (short)0, acc.v, false, false);
    }
  }

  // ---- epilogue (workspace rows padded to NP: unguarded for ws targets)
  const float sA = *pA, sB = *pB, sC = *pC, sD = *pD, sE = *pE;
#pragma unroll
  for (int t = 0; t < 2; t++) {
    FragC& acc = t ? acc1 : acc0;
    const int col = nBase + waveN * 32 + t * 16 + (lane & 15);
#pragma unroll
    for (int r = 0; r < 8; r++) {
      const int mrow = mBase + waveM * 16 + half * 8 + r;  // VGPR r: M=r / M=8+r
      if (MODE == 0) {
        float u = uaf(acc.f[r] + bias[col], sA, sB, sC, sD, sE);
        outH [mrow * HH + col] = u;
        outX0[mrow * HH + col] = u;
      } else {
        float S = 0.9f * Aa[mrow * HH + col] + 0.1f * Ab[mrow * HH + col];
        float v = (1.f - beta) * S + beta * acc.f[r];
        outH[mrow * HH + col] = uaf(v, sA, sB, sC, sD, sE);
      }
    }
  }
}

// ---------------------------------------------------------------------------
// Edge propagation: agg[dst] += h[src] * w.  4 edges per 256-thread block,
// float4 gather + per-float relaxed agent-scope fadd atomics (L2-resident).
// ---------------------------------------------------------------------------
__global__ void __launch_bounds__(256)
k_prop(const float* __restrict__ h, const int* __restrict__ ei,
       const float* __restrict__ ew, float* __restrict__ agg) {
  int tid = threadIdx.x;
  int e = blockIdx.x * 4 + (tid >> 6);          // E % 4 == 0, grid = E/4
  int src = ei[e];
  int dst = ei[EE + e];
  float w = ew[e];
  int f = (tid & 63) * 4;
  const float4 hv = *(const float4*)(h + src * HH + f);
  float* a = agg + dst * HH + f;
  __hip_atomic_fetch_add(a + 0, hv.x * w, __ATOMIC_RELAXED, __HIP_MEMORY_SCOPE_AGENT);
  __hip_atomic_fetch_add(a + 1, hv.y * w, __ATOMIC_RELAXED, __HIP_MEMORY_SCOPE_AGENT);
  __hip_atomic_fetch_add(a + 2, hv.z * w, __ATOMIC_RELAXED, __HIP_MEMORY_SCOPE_AGENT);
  __hip_atomic_fetch_add(a + 3, hv.w * w, __ATOMIC_RELAXED, __HIP_MEMORY_SCOPE_AGENT);
}

// ---------------------------------------------------------------------------
// Head: logits = h@W2 + b2, then log_softmax over 40 classes per node.
// W2 lives in LDS (40KB); one node per thread keeps all 40 logits in regs.
// ---------------------------------------------------------------------------
__global__ void __launch_bounds__(256)
k_head(const float* __restrict__ h, const float* __restrict__ W2,
       const float* __restrict__ b2, float* __restrict__ out) {
  __shared__ float w2[HH * CLS];
  __shared__ float bb[CLS];
  for (int i = threadIdx.x; i < HH * CLS; i += 256) w2[i] = W2[i];
  if (threadIdx.x < CLS) bb[threadIdx.x] = b2[threadIdx.x];
  __syncthreads();

  int node = blockIdx.x * 256 + threadIdx.x;
  if (node >= NN) return;

  float acc[CLS];
#pragma unroll
  for (int c = 0; c < CLS; c++) acc[c] = bb[c];

  const float* hr = h + node * HH;
  for (int k = 0; k < HH; k += 4) {
    float4 hv = *(const float4*)(hr + k);
#pragma unroll
    for (int c = 0; c < CLS; c++) {
      acc[c] += hv.x * w2[(k + 0) * CLS + c] + hv.y * w2[(k + 1) * CLS + c]
              + hv.z * w2[(k + 2) * CLS + c] + hv.w * w2[(k + 3) * CLS + c];
    }
  }
  float mx = acc[0];
#pragma unroll
  for (int c = 1; c < CLS; c++) mx = fmaxf(mx, acc[c]);
  float se = 0.f;
#pragma unroll
  for (int c = 0; c < CLS; c++) se += expf(acc[c] - mx);
  float lse = mx + logf(se);
#pragma unroll
  for (int c = 0; c < CLS; c++) out[node * CLS + c] = acc[c] - lse;
}

// ---------------------------------------------------------------------------
extern "C" void kernel_launch(void* const* d_in, const int* in_sizes, int n_in,
                              void* d_out, int out_size, void* d_ws, size_t ws_size,
                              hipStream_t stream) {
  const float* x  = (const float*)d_in[0];
  const int*   ei = (const int*)  d_in[1];
  const float* ew = (const float*)d_in[2];
  const float* W1 = (const float*)d_in[3];
  const float* b1 = (const float*)d_in[4];
  const float* W2 = (const float*)d_in[5];
  const float* b2 = (const float*)d_in[6];
  const float* Wc = (const float*)d_in[7];
  const float* pA = (const float*)d_in[8];
  const float* pB = (const float*)d_in[9];
  const float* pC = (const float*)d_in[10];
  const float* pD = (const float*)d_in[11];
  const float* pE = (const float*)d_in[12];

  char* ws = (char*)d_ws;
  size_t o = 0;
  float* hbuf = (float*)(ws + o);  o += (size_t)NP * HH * 4;   // padded rows
  float* x0   = (float*)(ws + o);  o += (size_t)NP * HH * 4;
  float* agg  = (float*)(ws + o);  o += (size_t)NP * HH * 4;
  unsigned short* W1bf = (unsigned short*)(ws + o);  o += (size_t)FF * HH * 2;
  unsigned short* Wcbf = (unsigned short*)(ws + o);  o += (size_t)LL * HH * HH * 2;

  // weights -> bf16 (max(F*H, L*H*H) = 262144 elems)
  k_cvt<<<(LL * HH * HH + 255) / 256, 256, 0, stream>>>(W1, Wc, W1bf, Wcbf);

  dim3 gg(NP / 64, HH / 64);

  // h = x0 = uaf(x @ W1 + b1)
  k_gemm<0><<<gg, 256, 0, stream>>>(x, nullptr, W1bf, b1, hbuf, x0,
                                    FF, 0.f, pA, pB, pC, pD, pE);

  for (int l = 0; l < LL; l++) {
    hipMemsetAsync(agg, 0, (size_t)NP * HH * 4, stream);
    k_prop<<<EE / 4, 256, 0, stream>>>(hbuf, ei, ew, agg);
    float beta = logf(0.5f / (float)(l + 1) + 1.f);
    k_gemm<1><<<gg, 256, 0, stream>>>(agg, x0, Wcbf + l * HH * HH, nullptr,
                                      hbuf, nullptr, HH, beta, pA, pB, pC, pD, pE);
  }

  k_head<<<(NN + 255) / 256, 256, 0, stream>>>(hbuf, W2, b2, (float*)d_out);
}
